// MultiHeadAttention_47596827574613
// MI455X (gfx1250) — matrix operand order repaired and checked by reading, not compiled, not run on hardware
//
#include <hip/hip_runtime.h>
#include <stdint.h>

// ---------------------------------------------------------------------------
// MI455X (gfx1250) multi-head attention: B=4, T=2048, D=1024, H=16, HD=64
// bf16 WMMA everywhere (v_wmma_f32_16x16x32_bf16), double-buffered async-to-LDS
// staging in the GEMMs, flash attention with online softmax.
// ---------------------------------------------------------------------------

typedef __bf16 bf16_t;
typedef __attribute__((ext_vector_type(16))) __bf16        v16bf;
typedef __attribute__((ext_vector_type(8)))  float         v8f;
typedef __attribute__((ext_vector_type(4)))  unsigned int  u32x4;

union Frag {            // one WMMA A/B operand: 16 bf16 per lane = 32 bytes
  u32x4 u4[2];
  v16bf v;
};

__device__ __forceinline__ bf16_t f2bf(float f) {
  union { float f; uint32_t u; } in; in.f = f;
  uint32_t r = in.u + 0x7FFFu + ((in.u >> 16) & 1u);   // round-nearest-even
  union { unsigned short s; bf16_t b; } out;
  out.s = (unsigned short)(r >> 16);
  return out.b;
}

// CDNA5 async copy global -> LDS (tracked on ASYNCcnt)
__device__ __forceinline__ void async_b128(uint32_t lds_off, uint32_t goff, uint64_t base) {
  asm volatile("global_load_async_to_lds_b128 %0, %1, %2"
               :: "v"(lds_off), "v"(goff), "s"(base) : "memory");
}
__device__ __forceinline__ void wait_async0() {
  asm volatile("s_wait_asynccnt 0" ::: "memory");
}

// ---------------------------------------------------------------------------
// fp32 -> bf16 elementwise cast
// ---------------------------------------------------------------------------
__global__ void mha_cast_bf16(const float* __restrict__ src, bf16_t* __restrict__ dst, int n) {
  int i = blockIdx.x * blockDim.x + threadIdx.x;
  if (i < n) dst[i] = f2bf(src[i]);
}

// ---------------------------------------------------------------------------
// C[8192,1024] = A[8192,1024] x W[1024,1024]^T  (i.e. sum_k A[m,k]*W[n,k])
// MODE 0: f32 out, linear [M][N]           (output projection -> d_out)
// MODE 1: bf16 out, [B,H,T,64] with scale  (Q with 1/sqrt(64), K with 1.0)
// MODE 2: bf16 out, [B,H,64,T] (V transposed for the PV WMMA B-fragments)
// Block: 256 thr = 8 waves, tile 128x128, K-step 32, double-buffered LDS via
// global_load_async_to_lds_b128 so tile i+1 streams while tile i runs WMMAs.
// ---------------------------------------------------------------------------
template <int MODE>
__global__ __launch_bounds__(256) void mha_gemm_bf16(
    const bf16_t* __restrict__ A, const bf16_t* __restrict__ W,
    void* __restrict__ out, float scale) {
  constexpr int K   = 1024;
  constexpr int LDA = 40;                 // padded row stride -> conflict-free b128
  constexpr int BUF = 128 * LDA;          // elems per stage buffer
  __shared__ __align__(16) bf16_t Abuf[2][BUF];
  __shared__ __align__(16) bf16_t Wbuf[2][BUF];

  const int tid  = threadIdx.x;
  const int lane = tid & 31;
  const int wid  = tid >> 5;
  const int wm   = wid >> 1;              // 0..3  (32-row wave tile)
  const int wn   = wid & 1;               // 0..1  (64-col wave tile)
  const int m0   = blockIdx.y * 128;
  const int n0   = blockIdx.x * 128;
  const int lm   = lane & 15;             // fragment row/col within 16
  const int lh   = lane >> 4;             // lane half

  const uint32_t aLds = (uint32_t)(size_t)&Abuf[0][0];
  const uint32_t wLds = (uint32_t)(size_t)&Wbuf[0][0];
  const uint64_t aG = (uint64_t)(size_t)A;
  const uint64_t wG = (uint64_t)(size_t)W;

  // Issue one 128x32 A tile + 128x32 W tile into stage `buf` (4 x b128/thread)
  auto issueTile = [&](int buf, int k0) {
    uint32_t lbase = (uint32_t)(buf * BUF * 2);
#pragma unroll
    for (int c = 0; c < 2; ++c) {
      int chunk = tid + 256 * c;          // 512 chunks of 16B each
      int row = chunk >> 2, cc = chunk & 3;
      uint32_t go = (uint32_t)(((size_t)(m0 + row) * K + k0 + cc * 8) * 2);
      async_b128(aLds + lbase + (uint32_t)(row * LDA * 2 + cc * 16), go, aG);
    }
#pragma unroll
    for (int c = 0; c < 2; ++c) {
      int chunk = tid + 256 * c;
      int row = chunk >> 2, cc = chunk & 3;
      uint32_t go = (uint32_t)(((size_t)(n0 + row) * K + k0 + cc * 8) * 2);
      async_b128(wLds + lbase + (uint32_t)(row * LDA * 2 + cc * 16), go, wG);
    }
  };

  v8f acc[2][4];
  v8f zero = {};
#pragma unroll
  for (int i = 0; i < 2; ++i)
#pragma unroll
    for (int j = 0; j < 4; ++j) acc[i][j] = zero;

  issueTile(0, 0);                        // prologue

  for (int k0 = 0, it = 0; k0 < K; k0 += 32, ++it) {
    const int buf = it & 1;
    wait_async0();                        // my async copies for stage `buf` done
    __syncthreads();                      // everyone's copies for `buf` done
    if (k0 + 32 < K) issueTile(buf ^ 1, k0 + 32);  // stream next tile under compute

    Frag fa[2], fb[4];
#pragma unroll
    for (int i = 0; i < 2; ++i) {         // A frags: rows = wm*32 + i*16 + lm
      const bf16_t* p = &Abuf[buf][(wm * 32 + i * 16 + lm) * LDA + 8 * lh];
      fa[i].u4[0] = *(const u32x4*)p;
      fa[i].u4[1] = *(const u32x4*)(p + 16);
    }
#pragma unroll
    for (int j = 0; j < 4; ++j) {         // B frags: cols = wn*64 + j*16 + lm
      const bf16_t* p = &Wbuf[buf][(wn * 64 + j * 16 + lm) * LDA + 8 * lh];
      fb[j].u4[0] = *(const u32x4*)p;
      fb[j].u4[1] = *(const u32x4*)(p + 16);
    }
#pragma unroll
    for (int i = 0; i < 2; ++i)
#pragma unroll
      for (int j = 0; j < 4; ++j)
        acc[i][j] = __builtin_amdgcn_wmma_f32_16x16x32_bf16(
            false, fa[i].v, false, fb[j].v, (short)0, acc[i][j], false, false);
  }

  // Epilogue: C-layout element (row = r + 8*lh, col = lm) per 16x16 frag
#pragma unroll
  for (int i = 0; i < 2; ++i)
#pragma unroll
    for (int j = 0; j < 4; ++j)
#pragma unroll
      for (int r = 0; r < 8; ++r) {
        int gm = m0 + wm * 32 + i * 16 + r + 8 * lh;
        int gn = n0 + wn * 64 + j * 16 + lm;
        float val = acc[i][j][r];
        if constexpr (MODE == 0) {
          ((float*)out)[(size_t)gm * 1024 + gn] = val;
        } else {
          int bb = gm >> 11, tt = gm & 2047;   // batch / time
          int hh = gn >> 6,  hd = gn & 63;     // head / head-dim
          bf16_t* obf = (bf16_t*)out;
          if constexpr (MODE == 1)
            obf[(((size_t)bb * 16 + hh) * 2048 + tt) * 64 + hd] = f2bf(val * scale);
          else  // V stored [B,H,64,T]
            obf[(((size_t)bb * 16 + hh) * 64 + hd) * 2048 + tt] = f2bf(val * scale);
        }
      }
}

// ---------------------------------------------------------------------------
// Causal flash attention. Grid (T/64, B*H), 128 thr = 4 waves x 16 q-rows.
// q,k: [B,H,T,64] bf16 (q pre-scaled by 1/8), v: [B,H,64,T] bf16.
// ctx out: [B,T,D] bf16 with D = h*64+hd (matches reference transpose/reshape)
// ---------------------------------------------------------------------------
__global__ __launch_bounds__(128) void mha_flash_attn(
    const bf16_t* __restrict__ q, const bf16_t* __restrict__ k,
    const bf16_t* __restrict__ v, bf16_t* __restrict__ ctx) {
  constexpr int T = 2048, HD = 64, LDP = 40;
  __shared__ __align__(16) bf16_t Pbuf[4 * 16 * LDP];  // wave-private P tiles

  const int bh   = blockIdx.y;
  const int b    = bh >> 4, h = bh & 15;
  const int lane = threadIdx.x & 31;
  const int wid  = threadIdx.x >> 5;
  const int qr0  = blockIdx.x * 64 + wid * 16;
  const int lm   = lane & 15, lh = lane >> 4;

  const bf16_t* qb = q + (size_t)bh * T * HD;
  const bf16_t* kb = k + (size_t)bh * T * HD;
  const bf16_t* vb = v + (size_t)bh * HD * T;

  // Q A-fragments for K-dim chunks [0,32) and [32,64)
  Frag aq[2];
#pragma unroll
  for (int f = 0; f < 2; ++f) {
    const bf16_t* p = qb + (size_t)(qr0 + lm) * HD + 32 * f + 8 * lh;
    aq[f].u4[0] = *(const u32x4*)p;
    aq[f].u4[1] = *(const u32x4*)(p + 16);
  }

  float mrow[8], lrow[8];
  v8f acc[4];
  v8f zero = {};
#pragma unroll
  for (int r = 0; r < 8; ++r) { mrow[r] = -1e30f; lrow[r] = 0.f; }
#pragma unroll
  for (int t = 0; t < 4; ++t) acc[t] = zero;

  bf16_t* P = &Pbuf[wid * 16 * LDP];

  for (int kv0 = 0; kv0 < qr0 + 16; kv0 += 32) {
    // S = Q * K^T : two 16x16 column tiles, K-dim 64 = 2 WMMA steps each
    v8f s[2];
#pragma unroll
    for (int t = 0; t < 2; ++t) {
      v8f c = zero;
#pragma unroll
      for (int f = 0; f < 2; ++f) {
        Frag bk;  // B-frag: lane = key column, rows = hd
        const bf16_t* p = kb + (size_t)(kv0 + 16 * t + lm) * HD + 32 * f + 8 * lh;
        bk.u4[0] = *(const u32x4*)p;
        bk.u4[1] = *(const u32x4*)(p + 16);
        c = __builtin_amdgcn_wmma_f32_16x16x32_bf16(
            false, aq[f].v, false, bk.v, (short)0, c, false, false);
      }
      s[t] = c;
    }

    // Causal mask (only diagonal-crossing tiles need it)
    if (kv0 + 31 > qr0) {
#pragma unroll
      for (int t = 0; t < 2; ++t)
#pragma unroll
        for (int r = 0; r < 8; ++r)
          if (kv0 + 16 * t + lm > qr0 + r + 8 * lh) s[t][r] = -1e30f;
    }

    // Online softmax per row (rows live in lane halves; reduce across 16 lanes)
#pragma unroll
    for (int r = 0; r < 8; ++r) {
      float mx = fmaxf(s[0][r], s[1][r]);
#pragma unroll
      for (int off = 1; off < 16; off <<= 1) mx = fmaxf(mx, __shfl_xor(mx, off, 32));
      float mnew = fmaxf(mrow[r], mx);
      float corr = __expf(mrow[r] - mnew);
      float p0 = __expf(s[0][r] - mnew);
      float p1 = __expf(s[1][r] - mnew);
      float ps = p0 + p1;
#pragma unroll
      for (int off = 1; off < 16; off <<= 1) ps += __shfl_xor(ps, off, 32);
      lrow[r] = lrow[r] * corr + ps;
      mrow[r] = mnew;
#pragma unroll
      for (int t = 0; t < 4; ++t) acc[t][r] *= corr;
      // C-layout -> LDS [row][kv] bf16 (wave-private; DS pipe is in-order)
      P[(r + 8 * lh) * LDP + lm]      = f2bf(p0);
      P[(r + 8 * lh) * LDP + 16 + lm] = f2bf(p1);
    }

    // Reload P as an A-fragment (16x32) and do PV WMMAs over 4 hd tiles
    Frag ap;
    {
      const bf16_t* pp = P + lm * LDP + 8 * lh;
      ap.u4[0] = *(const u32x4*)pp;
      ap.u4[1] = *(const u32x4*)(pp + 16);
    }
#pragma unroll
    for (int t = 0; t < 4; ++t) {
      Frag bv;  // B-frag from transposed V: lane = hd column, rows = kv
      const bf16_t* p = vb + (size_t)(16 * t + lm) * T + kv0 + 8 * lh;
      bv.u4[0] = *(const u32x4*)p;
      bv.u4[1] = *(const u32x4*)(p + 16);
      acc[t] = __builtin_amdgcn_wmma_f32_16x16x32_bf16(
          false, ap.v, false, bv.v, (short)0, acc[t], false, false);
    }
  }

  // Normalize and store ctx[b][t][h*64+hd]
  bf16_t* cb = ctx + ((size_t)b * 2048) * 1024 + (size_t)h * 64;
#pragma unroll
  for (int t = 0; t < 4; ++t)
#pragma unroll
    for (int r = 0; r < 8; ++r) {
      int row = qr0 + r + 8 * lh;
      int hd  = 16 * t + lm;
      cb[(size_t)row * 1024 + hd] = f2bf(acc[t][r] / lrow[r]);
    }
}

// ---------------------------------------------------------------------------
extern "C" void kernel_launch(void* const* d_in, const int* in_sizes, int n_in,
                              void* d_out, int out_size, void* d_ws, size_t ws_size,
                              hipStream_t stream) {
  (void)in_sizes; (void)n_in; (void)out_size; (void)ws_size;
  const float* x  = (const float*)d_in[0];
  const float* Wq = (const float*)d_in[1];
  const float* Wk = (const float*)d_in[2];
  const float* Wv = (const float*)d_in[3];
  const float* Wp = (const float*)d_in[4];

  constexpr size_t NX = 8192ull * 1024;   // x / q / k / v / ctx element counts
  constexpr size_t NW = 1024ull * 1024;

  bf16_t* ws    = (bf16_t*)d_ws;
  bf16_t* x_bf  = ws;  ws += NX;
  bf16_t* wq_bf = ws;  ws += NW;
  bf16_t* wk_bf = ws;  ws += NW;
  bf16_t* wv_bf = ws;  ws += NW;
  bf16_t* wp_bf = ws;  ws += NW;
  bf16_t* q_bf  = ws;  ws += NX;
  bf16_t* k_bf  = ws;  ws += NX;
  bf16_t* v_bf  = ws;  ws += NX;
  bf16_t* c_bf  = ws;  ws += NX;

  mha_cast_bf16<<<(int)(NX / 256), 256, 0, stream>>>(x,  x_bf,  (int)NX);
  mha_cast_bf16<<<(int)(NW / 256), 256, 0, stream>>>(Wq, wq_bf, (int)NW);
  mha_cast_bf16<<<(int)(NW / 256), 256, 0, stream>>>(Wk, wk_bf, (int)NW);
  mha_cast_bf16<<<(int)(NW / 256), 256, 0, stream>>>(Wv, wv_bf, (int)NW);
  mha_cast_bf16<<<(int)(NW / 256), 256, 0, stream>>>(Wp, wp_bf, (int)NW);

  dim3 gg(8, 64);  // N/128, M/128
  mha_gemm_bf16<1><<<gg, 256, 0, stream>>>(x_bf, wq_bf, q_bf, 0.125f); // Q * 1/sqrt(64)
  mha_gemm_bf16<1><<<gg, 256, 0, stream>>>(x_bf, wk_bf, k_bf, 1.0f);
  mha_gemm_bf16<2><<<gg, 256, 0, stream>>>(x_bf, wv_bf, v_bf, 1.0f);   // V transposed

  mha_flash_attn<<<dim3(32, 64), 128, 0, stream>>>(q_bf, k_bf, v_bf, c_bf);

  mha_gemm_bf16<0><<<gg, 256, 0, stream>>>(c_bf, wp_bf, d_out, 1.0f);
}
